// SparseFocalModulation_627065225786
// MI455X (gfx1250) — matrix operand order repaired and checked by reading, not compiled, not run on hardware
//
#include <hip/hip_runtime.h>
#include <math.h>

// ---------------- constants ----------------
#define NROWS   30000
#define DIMC    64
#define NTILES  1875        // NROWS / 16
#define GBLK    235         // ceil(NTILES / 8)

typedef __attribute__((ext_vector_type(16))) _Float16 v16h;
typedef __attribute__((ext_vector_type(8)))  float    v8f;

union AB16 {
    v16h  v;
    uint4 q[2];
    _Float16 e[16];
};

// Load A fragment (16x32 f16, row-major src, ld=64) for WMMA 16x16x32.
// Lane<16 holds row m=lane, K = kbase+[0..8) and kbase+[16..24);
// Lane>=16 holds row m=lane-16, K = kbase+[8..16) and kbase+[24..32).
__device__ __forceinline__ void load_a_frag(AB16& f, const _Float16* src, int row,
                                            int kbase, int lane) {
    const int k0 = kbase + ((lane & 16) ? 8 : 0);
    const _Float16* p = src + (size_t)row * DIMC + k0;
    f.q[0] = *(const uint4*)(p);
    f.q[1] = *(const uint4*)(p + 16);
}

// Load B fragment (32x16 f16) from transposed weights Wt[n][k] (ld=64).
// Lane<16: col n = nbase+lane, K = kbase+[0..8) and kbase+[16..24); lane>=16: same col, K = kbase+[8..16) and kbase+[24..32).
__device__ __forceinline__ void load_b_frag(AB16& f, const _Float16* wt, int nbase,
                                            int kbase, int lane) {
    const int col = nbase + (lane & 15);
    const int k0  = kbase + ((lane & 16) ? 8 : 0);
    const _Float16* p = wt + (size_t)col * DIMC + k0;
    f.q[0] = *(const uint4*)(p);
    f.q[1] = *(const uint4*)(p + 16);
}

__device__ __forceinline__ void zero_frag(AB16& f) {
    f.q[0] = make_uint4(0u, 0u, 0u, 0u);
    f.q[1] = make_uint4(0u, 0u, 0u, 0u);
}

__device__ __forceinline__ float gelu_exact(float x) {
    return 0.5f * x * (1.0f + erff(x * 0.70710678118654752f));
}

// ---------------- prep kernels ----------------
__global__ void cvt_f16_kernel(const float* __restrict__ src, _Float16* __restrict__ dst, int n) {
    int i = blockIdx.x * 256 + threadIdx.x;
    if (i < n) dst[i] = (_Float16)src[i];
}

// src: [nblk][64(k)][64(n)] f32 -> dst: [nblk][64(n)][64(k)] f16
__global__ void transpose64_kernel(const float* __restrict__ src, _Float16* __restrict__ dst, int nblk) {
    int i = blockIdx.x * 256 + threadIdx.x;
    if (i >= nblk * 4096) return;
    int o = i >> 12, r = i & 4095, n = r >> 6, k = r & 63;
    dst[i] = (_Float16)src[(o << 12) + k * DIMC + n];
}

// f_W: [64(k)][132(n)] f32 -> f_Wt: [144(n)][64(k)] f16, zero-padded cols 132..143
__global__ void transpose_f_kernel(const float* __restrict__ src, _Float16* __restrict__ dst) {
    int i = blockIdx.x * 256 + threadIdx.x;
    if (i >= 144 * DIMC) return;
    int n = i >> 6, k = i & 63;
    dst[i] = (n < 132) ? (_Float16)src[k * 132 + n] : (_Float16)0.0f;
}

__global__ void zero_kernel(float* __restrict__ p, int n) {
    int i = blockIdx.x * 256 + threadIdx.x;
    if (i < n) p[i] = 0.0f;
}

// ---------------- xf = features @ f_W + f_b ; split q/ctx/gates ----------------
__global__ void __launch_bounds__(256) gemm_f_kernel(const _Float16* __restrict__ A,
                                                     const _Float16* __restrict__ Bt,
                                                     const float* __restrict__ fb,
                                                     float* __restrict__ qout,
                                                     _Float16* __restrict__ ctx16,
                                                     float* __restrict__ gates) {
    const int lane = threadIdx.x & 31;
    const int tile = blockIdx.x * 8 + (threadIdx.x >> 5);
    if (tile >= NTILES) return;
    const int r0 = tile * 16;
    const int m = lane & 15;

    AB16 a0, a1;
    load_a_frag(a0, A, r0 + m, 0, lane);
    load_a_frag(a1, A, r0 + m, 32, lane);

    const int rbase = r0 + ((lane & 16) ? 8 : 0);
    for (int nt = 0; nt < 9; ++nt) {
        v8f acc = {};
        AB16 b;
        load_b_frag(b, Bt, nt * 16, 0, lane);
        acc = __builtin_amdgcn_wmma_f32_16x16x32_f16(false, a0.v, false, b.v, (short)0, acc, false, false);
        load_b_frag(b, Bt, nt * 16, 32, lane);
        acc = __builtin_amdgcn_wmma_f32_16x16x32_f16(false, a1.v, false, b.v, (short)0, acc, false, false);

        const int col = nt * 16 + m;
        if (col < 132) {
            const float bias = fb[col];
#pragma unroll
            for (int e = 0; e < 8; ++e) {
                const int row = rbase + e;
                const float v = acc[e] + bias;
                if (col < 64)       qout[row * DIMC + col] = v;
                else if (col < 128) ctx16[row * DIMC + (col - 64)] = (_Float16)v;
                else                gates[row * 4 + (col - 128)] = v;
            }
        }
    }
}

// ---------------- submanifold conv, offset-split with atomic accumulation ----------------
// out[i] += sum_{o in chunk} gather(ctx, idx[o,i]) @ W[o]   (bias applied later in LN)
__global__ void __launch_bounds__(256) subm_conv_kernel(const _Float16* __restrict__ ctx16,
                                                        const int* __restrict__ idx,
                                                        const _Float16* __restrict__ wt,
                                                        float* __restrict__ out,
                                                        int K, int ochunk) {
    __shared__ _Float16 Wl[2][4096];   // double-buffered 64x64 f16 weight tile, [n][k]
    const int t    = threadIdx.x;
    const int lane = t & 31;
    const int tile = blockIdx.x * 8 + (t >> 5);
    const int r0   = tile * 16;
    const bool valid = (r0 < NROWS);
    const int m = lane & 15;
    const int rload = valid ? (r0 + m) : 0;

    const int o0   = blockIdx.y * ochunk;
    const int oend = (o0 + ochunk < K) ? (o0 + ochunk) : K;

    v8f acc[4] = {};

    // ---- prologue: weight regs + A fragments for first offset ----
    uint4 w0 = ((const uint4*)(wt + (size_t)o0 * 4096))[t];
    uint4 w1 = ((const uint4*)(wt + (size_t)o0 * 4096))[t + 256];
    AB16 a0c, a1c;
    {
        const int nb = idx[(size_t)o0 * NROWS + rload];
        if (nb >= 0) {
            load_a_frag(a0c, ctx16, nb, 0, lane);
            load_a_frag(a1c, ctx16, nb, 32, lane);
        } else { zero_frag(a0c); zero_frag(a1c); }
    }

    for (int o = o0; o < oend; ++o) {
        const int cur = (o - o0) & 1;
        // stage current weights (regs were fetched one iteration ago)
        {
            uint4* d = (uint4*)&Wl[cur][0];
            d[t]       = w0;
            d[t + 256] = w1;
        }
        __syncthreads();   // Wl[cur] visible; previous iteration's reads of Wl[cur^1] done

        // issue next-offset weight fetch + A gather; consumed only after the WMMAs
        AB16 a0n, a1n;
        if (o + 1 < oend) {
            const uint4* s = (const uint4*)(wt + (size_t)(o + 1) * 4096);
            w0 = s[t];
            w1 = s[t + 256];
            if (o + 2 < oend)
                __builtin_prefetch(wt + (size_t)(o + 2) * 4096 + t * 16, 0, 0);
            const int nb = idx[(size_t)(o + 1) * NROWS + rload];
            if (nb >= 0) {
                load_a_frag(a0n, ctx16, nb, 0, lane);
                load_a_frag(a1n, ctx16, nb, 32, lane);
            } else { zero_frag(a0n); zero_frag(a1n); }
        } else { zero_frag(a0n); zero_frag(a1n); }

#pragma unroll
        for (int nt = 0; nt < 4; ++nt) {
            AB16 b;
            load_b_frag(b, &Wl[cur][0], nt * 16, 0, lane);
            acc[nt] = __builtin_amdgcn_wmma_f32_16x16x32_f16(false, a0c.v, false, b.v, (short)0, acc[nt], false, false);
            load_b_frag(b, &Wl[cur][0], nt * 16, 32, lane);
            acc[nt] = __builtin_amdgcn_wmma_f32_16x16x32_f16(false, a1c.v, false, b.v, (short)0, acc[nt], false, false);
        }
        a0c = a0n;
        a1c = a1n;
    }

    if (valid) {
        const int rbase = r0 + ((lane & 16) ? 8 : 0);
#pragma unroll
        for (int nt = 0; nt < 4; ++nt) {
            const int col = nt * 16 + m;
#pragma unroll
            for (int e = 0; e < 8; ++e)
                unsafeAtomicAdd(&out[(size_t)(rbase + e) * DIMC + col], acc[nt][e]);
        }
    }
}

// ---------------- bias + gelu + LayerNorm + gated accumulate (+ pool atomics on last level) ----------------
__global__ void __launch_bounds__(256) ln_gate_kernel(const float* __restrict__ cin,
                                                      const float* __restrict__ cbias,
                                                      const float* __restrict__ g,
                                                      const float* __restrict__ be,
                                                      const float* __restrict__ gates,
                                                      const int* __restrict__ bids,
                                                      float* __restrict__ ctx_all,
                                                      _Float16* __restrict__ ctx16,
                                                      float* __restrict__ pool,
                                                      int level) {
    const int lane = threadIdx.x & 31;
    const int row  = blockIdx.x * 8 + (threadIdx.x >> 5);
    if (row >= NROWS) return;

    const float x0 = gelu_exact(cin[(size_t)row * DIMC + lane]      + cbias[lane]);
    const float x1 = gelu_exact(cin[(size_t)row * DIMC + 32 + lane] + cbias[32 + lane]);

    float s = x0 + x1;
#pragma unroll
    for (int d = 16; d >= 1; d >>= 1) s += __shfl_xor(s, d, 32);
    const float mean = s * (1.0f / 64.0f);
    const float d0 = x0 - mean, d1 = x1 - mean;
    float var = d0 * d0 + d1 * d1;
#pragma unroll
    for (int d = 16; d >= 1; d >>= 1) var += __shfl_xor(var, d, 32);
    const float rinv = rsqrtf(var * (1.0f / 64.0f) + 1e-5f);

    const float y0 = d0 * rinv * g[lane]      + be[lane];
    const float y1 = d1 * rinv * g[32 + lane] + be[32 + lane];

    ctx16[(size_t)row * DIMC + lane]      = (_Float16)y0;
    ctx16[(size_t)row * DIMC + 32 + lane] = (_Float16)y1;

    const float gt = gates[row * 4 + level];
    float* ca = ctx_all + (size_t)row * DIMC;
    if (level == 0) {
        ca[lane]      = y0 * gt;
        ca[32 + lane] = y1 * gt;
    } else {
        ca[lane]      += y0 * gt;
        ca[32 + lane] += y1 * gt;
    }
    if (level == 2) {
        const int b = bids[row];
        atomicAdd(&pool[b * DIMC + lane], y0);
        atomicAdd(&pool[b * DIMC + 32 + lane], y1);
        if (lane == 0) atomicAdd(&pool[256 + b], 1.0f);
    }
}

// pool layout: [0..255] sums, [256..259] counts, [260..515] gelu(mean)
__global__ void pool_fin_kernel(float* __restrict__ pool) {
    const int t = threadIdx.x;   // 256 threads
    const int b = t >> 6;
    pool[260 + t] = gelu_exact(pool[t] / pool[256 + b]);
}

__global__ void add_pool_kernel(const float* __restrict__ ctx_all,
                                const float* __restrict__ pooled,
                                const int* __restrict__ bids,
                                const float* __restrict__ gates,
                                _Float16* __restrict__ ctxall16) {
    const int i = blockIdx.x * 256 + threadIdx.x;
    if (i >= NROWS * DIMC) return;
    const int row = i >> 6, c = i & 63;
    const float v = ctx_all[i] + pooled[bids[row] * DIMC + c] * gates[row * 4 + 3];
    ctxall16[i] = (_Float16)v;
}

// ---------------- t = ctx_all @ h_W + h_b ; u = q * t ----------------
__global__ void __launch_bounds__(256) gemm_h_kernel(const _Float16* __restrict__ A,
                                                     const _Float16* __restrict__ Bt,
                                                     const float* __restrict__ hb,
                                                     const float* __restrict__ qin,
                                                     _Float16* __restrict__ u16) {
    const int lane = threadIdx.x & 31;
    const int tile = blockIdx.x * 8 + (threadIdx.x >> 5);
    if (tile >= NTILES) return;
    const int r0 = tile * 16;
    const int m = lane & 15;

    AB16 a0, a1;
    load_a_frag(a0, A, r0 + m, 0, lane);
    load_a_frag(a1, A, r0 + m, 32, lane);

    const int rbase = r0 + ((lane & 16) ? 8 : 0);
#pragma unroll
    for (int nt = 0; nt < 4; ++nt) {
        v8f acc = {};
        AB16 b;
        load_b_frag(b, Bt, nt * 16, 0, lane);
        acc = __builtin_amdgcn_wmma_f32_16x16x32_f16(false, a0.v, false, b.v, (short)0, acc, false, false);
        load_b_frag(b, Bt, nt * 16, 32, lane);
        acc = __builtin_amdgcn_wmma_f32_16x16x32_f16(false, a1.v, false, b.v, (short)0, acc, false, false);

        const int col = nt * 16 + m;
        const float bias = hb[col];
#pragma unroll
        for (int e = 0; e < 8; ++e) {
            const int row = rbase + e;
            const float tv = acc[e] + bias;
            u16[(size_t)row * DIMC + col] = (_Float16)(qin[(size_t)row * DIMC + col] * tv);
        }
    }
}

// ---------------- out = u @ proj_W + proj_b ----------------
__global__ void __launch_bounds__(256) gemm_proj_kernel(const _Float16* __restrict__ A,
                                                        const _Float16* __restrict__ Bt,
                                                        const float* __restrict__ pb,
                                                        float* __restrict__ out) {
    const int lane = threadIdx.x & 31;
    const int tile = blockIdx.x * 8 + (threadIdx.x >> 5);
    if (tile >= NTILES) return;
    const int r0 = tile * 16;
    const int m = lane & 15;

    AB16 a0, a1;
    load_a_frag(a0, A, r0 + m, 0, lane);
    load_a_frag(a1, A, r0 + m, 32, lane);

    const int rbase = r0 + ((lane & 16) ? 8 : 0);
#pragma unroll
    for (int nt = 0; nt < 4; ++nt) {
        v8f acc = {};
        AB16 b;
        load_b_frag(b, Bt, nt * 16, 0, lane);
        acc = __builtin_amdgcn_wmma_f32_16x16x32_f16(false, a0.v, false, b.v, (short)0, acc, false, false);
        load_b_frag(b, Bt, nt * 16, 32, lane);
        acc = __builtin_amdgcn_wmma_f32_16x16x32_f16(false, a1.v, false, b.v, (short)0, acc, false, false);

        const int col = nt * 16 + m;
        const float bias = pb[col];
#pragma unroll
        for (int e = 0; e < 8; ++e)
            out[(size_t)(rbase + e) * DIMC + col] = acc[e] + bias;
    }
}

// ---------------- host ----------------
extern "C" void kernel_launch(void* const* d_in, const int* in_sizes, int n_in,
                              void* d_out, int out_size, void* d_ws, size_t ws_size,
                              hipStream_t stream) {
    (void)in_sizes; (void)n_in; (void)out_size; (void)ws_size;
    const float* features = (const float*)d_in[0];
    const float* f_W      = (const float*)d_in[1];
    const float* f_b      = (const float*)d_in[2];
    const float* h_W      = (const float*)d_in[3];
    const float* h_b      = (const float*)d_in[4];
    const float* proj_W   = (const float*)d_in[5];
    const float* proj_b   = (const float*)d_in[6];
    const int*   bids     = (const int*)  d_in[7];
    const float* wl[3]  = {(const float*)d_in[8],  (const float*)d_in[13], (const float*)d_in[18]};
    const float* bl[3]  = {(const float*)d_in[9],  (const float*)d_in[14], (const float*)d_in[19]};
    const float* gl[3]  = {(const float*)d_in[10], (const float*)d_in[15], (const float*)d_in[20]};
    const float* bel[3] = {(const float*)d_in[11], (const float*)d_in[16], (const float*)d_in[21]};
    const int*   idxl[3]= {(const int*)  d_in[12], (const int*)  d_in[17], (const int*)  d_in[22]};
    const int Kl[3]    = {27, 125, 343};
    const int Ksplit[3] = {3, 5, 7};     // exact divisors -> uniform chunks

    char* wp = (char*)d_ws;
    auto alloc = [&](size_t bytes) -> void* {
        void* r = (void*)wp;
        wp += (bytes + 255) & ~(size_t)255;
        return r;
    };
    _Float16* feat16   = (_Float16*)alloc((size_t)NROWS * DIMC * 2);
    _Float16* fWt      = (_Float16*)alloc(144 * DIMC * 2);
    _Float16* hWt      = (_Float16*)alloc(4096 * 2);
    _Float16* pWt      = (_Float16*)alloc(4096 * 2);
    _Float16* Wt[3];
    for (int l = 0; l < 3; ++l) Wt[l] = (_Float16*)alloc((size_t)Kl[l] * 4096 * 2);
    float*    qws      = (float*)alloc((size_t)NROWS * DIMC * 4);
    float*    gatesws  = (float*)alloc((size_t)NROWS * 4 * 4);
    _Float16* ctx16    = (_Float16*)alloc((size_t)NROWS * DIMC * 2);
    float*    convout  = (float*)alloc((size_t)NROWS * DIMC * 4);
    float*    ctxall   = (float*)alloc((size_t)NROWS * DIMC * 4);
    float*    pool     = (float*)alloc(516 * 4);
    _Float16* ctxall16 = (_Float16*)alloc((size_t)NROWS * DIMC * 2);
    _Float16* u16      = (_Float16*)alloc((size_t)NROWS * DIMC * 2);

    // prep
    zero_kernel<<<2, 256, 0, stream>>>(pool, 260);
    cvt_f16_kernel<<<(NROWS * DIMC + 255) / 256, 256, 0, stream>>>(features, feat16, NROWS * DIMC);
    transpose_f_kernel<<<(144 * DIMC + 255) / 256, 256, 0, stream>>>(f_W, fWt);
    transpose64_kernel<<<16, 256, 0, stream>>>(h_W, hWt, 1);
    transpose64_kernel<<<16, 256, 0, stream>>>(proj_W, pWt, 1);
    for (int l = 0; l < 3; ++l)
        transpose64_kernel<<<(Kl[l] * 4096 + 255) / 256, 256, 0, stream>>>(wl[l], Wt[l], Kl[l]);

    // xf = features @ f_W + f_b -> q, ctx, gates
    gemm_f_kernel<<<GBLK, 256, 0, stream>>>(feat16, fWt, f_b, qws, ctx16, gatesws);

    // focal levels
    for (int l = 0; l < 3; ++l) {
        zero_kernel<<<(NROWS * DIMC + 255) / 256, 256, 0, stream>>>(convout, NROWS * DIMC);
        dim3 grid(GBLK, Ksplit[l]);
        subm_conv_kernel<<<grid, 256, 0, stream>>>(ctx16, idxl[l], Wt[l], convout,
                                                   Kl[l], Kl[l] / Ksplit[l]);
        ln_gate_kernel<<<(NROWS + 7) / 8, 256, 0, stream>>>(convout, bl[l], gl[l], bel[l], gatesws,
                                                            bids, ctxall, ctx16, pool, l);
    }

    // pooled context
    pool_fin_kernel<<<1, 256, 0, stream>>>(pool);
    add_pool_kernel<<<(NROWS * DIMC + 255) / 256, 256, 0, stream>>>(ctxall, pool + 260, bids,
                                                                    gatesws, ctxall16);

    // ctx_all @ h_W + h_b ; * q ; @ proj_W + proj_b
    gemm_h_kernel<<<GBLK, 256, 0, stream>>>(ctxall16, hWt, h_b, qws, u16);
    gemm_proj_kernel<<<GBLK, 256, 0, stream>>>(u16, pWt, proj_b, (float*)d_out);
}
